// BlobLayer_24266565222902
// MI455X (gfx1250) — matrix operand order repaired and checked by reading, not compiled, not run on hardware
//
#include <hip/hip_runtime.h>

// Problem constants (from reference)
#define HH 224
#define WW 224
#define CC 1024
#define BB 256
#define KTOT (HH * WW)          // 50176 pixels (K dimension)
#define ROWSPB 4                // image rows per K-split block
#define KCHUNK (ROWSPB * WW)    // 896 pixels per block
#define KSPLIT (HH / ROWSPB)    // 56 K-split blocks
#define NB 32                   // channels per block (2 N-tiles)
#define NBLK (CC / NB)          // 32 channel blocks
#define PITER (2 * WW)          // 448 pixels (2 rows) staged per iteration
#define NIT (KCHUNK / PITER)    // 2 outer iterations
#define NSUB (PITER / 32)       // 14 WMMA K-steps per iteration
#define BPAD 456                // LDS pitch in halfs: 16B-aligned, 36 words mod 64 (bank-spread)
#define EPS_F 0.001f
#define CAP_F 2000.0f
#define TWO_PI_F 6.283185307179586f
#define INV224 (1.0f / 224.0f)

typedef __attribute__((ext_vector_type(16))) _Float16 v16h;
typedef __attribute__((ext_vector_type(8)))  float    v8f;
typedef __attribute__((ext_vector_type(4)))  unsigned int u32x4;
typedef __attribute__((ext_vector_type(4)))  float    f32x4;
typedef __attribute__((ext_vector_type(4)))  _Float16 f16x4;

__device__ __forceinline__ unsigned int pack2h(float a, float b) {
  union { _Float16 h[2]; unsigned int u; } p;
  p.h[0] = (_Float16)a;
  p.h[1] = (_Float16)b;
  return p.u;
}

// ---------------------------------------------------------------------------
// Kernel 1: cast x (f32) -> f16 workspace copy (L2-resident working set).
// ---------------------------------------------------------------------------
__global__ __launch_bounds__(256) void xcast_f16(const float* __restrict__ x,
                                                 _Float16* __restrict__ xh,
                                                 int n4) {
  int i = blockIdx.x * blockDim.x + threadIdx.x;
  if (i < n4) {
    f32x4 v = *(const f32x4*)(x + (size_t)i * 4);
    f16x4 h;
    h.x = (_Float16)v.x; h.y = (_Float16)v.y;
    h.z = (_Float16)v.z; h.w = (_Float16)v.w;
    *(f16x4*)(xh + (size_t)i * 4) = h;
  }
}

// ---------------------------------------------------------------------------
// Kernel 2: zero output accumulator (harness poisons d_out with 0xAA).
// ---------------------------------------------------------------------------
__global__ __launch_bounds__(256) void zero_out(float* __restrict__ out, int n) {
  int i = blockIdx.x * blockDim.x + threadIdx.x;
  if (i < n) out[i] = 0.0f;
}

// ---------------------------------------------------------------------------
// Kernel 3: fused curve-generation + f16 WMMA GEMM, row-aligned split-K.
//   grid.x = NBLK * KSPLIT = 1792 blocks of 256 threads (8 waves).
//   Block (nt, ks): channels [nt*32, nt*32+32), image rows [ks*4, ks*4+4).
//   Each wave: 32 batch rows x 32 channels = 2x2 register tile of 16x16 accs.
//   Double-buffered 2-row LDS curve tile; 56 WMMAs/wave per barrier.
// ---------------------------------------------------------------------------
__global__ __launch_bounds__(256) void blob_wmma(
    const _Float16* __restrict__ xh,   // [BB][KTOT] f16
    const float* __restrict__ pos,     // [CC][2] (idx0=y, idx1=x)
    const float* __restrict__ sig,     // [CC]
    const float* __restrict__ cw,      // [CC]
    float* __restrict__ out)           // [BB][CC] f32, atomically accumulated
{
  __shared__ __align__(16) _Float16 sB[2][NB][BPAD];  // [buf][channel][pixel-in-window]

  const int nt    = blockIdx.x % NBLK;
  const int ks    = blockIdx.x / NBLK;
  const int c0    = nt * NB;
  const int rbase = ks * ROWSPB;            // first image row of this block
  const int kbase = rbase * WW;             // first pixel (flat K index)

  const int t    = threadIdx.x;
  const int lane = t & 31;
  const int wv   = t >> 5;                  // 0..7 -> batch rows [wv*32, wv*32+32)

  // ---- curve-generation assignment: channel t>>3; 56 pixels of one image row ----
  const int   cgl  = t >> 3;                // local channel 0..31
  const int   g    = t & 7;
  const int   rloc = g >> 2;                // row within 2-row window (0/1)
  const int   w0   = (g & 3) * 56;          // first column of this thread's span
  const int   cch  = c0 + cgl;
  const float cpy  = pos[2 * cch + 0];
  const float cpx  = pos[2 * cch + 1];
  const float css  = sig[cch] * sig[cch];
  const float cinv = 1.0f / (2.0f * css + EPS_F);
  const float cfac = cw[cch] / (TWO_PI_F * css + EPS_F);

  // ---- WMMA fragment addressing ----
  // B (32x16 f16): lanes 0-15 -> K 0..15, lanes 16-31 -> K 16..31 (contiguous k order, same as A)
  const int bn   = lane & 15;
  const int bcol = (lane >> 4) * 16;
  // A (16x32 f16): lanes 0-15 -> K 0..15, lanes 16-31 -> K 16..31 (matches B)
  const int am = lane & 15;
  const int ak = (lane >> 4) * 16;
  const size_t row0 = (size_t)(wv * 32 + am) * KTOT;
  const size_t row1 = row0 + (size_t)16 * KTOT;

  v8f acc00 = {}, acc01 = {}, acc10 = {}, acc11 = {};

  // ---- curve tile generator: 56 same-row pixels for channel cgl into sB[buf] ----
  auto gen_curves = [&](int it, int buf) {
    const int   irow = rbase + it * 2 + rloc;        // image row (constant y)
    const float fy   = (float)irow * INV224;
    const float dy   = fy - cpy;
    const float cexp = dy * dy * cinv;               // row-constant exponent term
    u32x4* dst = (u32x4*)&sB[buf][cgl][rloc * WW + w0];
#pragma unroll
    for (int q = 0; q < 7; ++q) {
      unsigned int pk[4];
#pragma unroll
      for (int jj = 0; jj < 4; ++jj) {
        int w = w0 + 8 * q + 2 * jj;
        float dx0 = (float)w * INV224 - cpx;
        float dx1 = (float)(w + 1) * INV224 - cpx;
        float v0 = cfac * __expf(-(dx0 * dx0 * cinv + cexp));
        float v1 = cfac * __expf(-(dx1 * dx1 * cinv + cexp));
        v0 = fminf(fmaxf(v0, -CAP_F), CAP_F);
        v1 = fminf(fmaxf(v1, -CAP_F), CAP_F);
        pk[jj] = pack2h(v0, v1);
      }
      u32x4 d = {pk[0], pk[1], pk[2], pk[3]};
      dst[q] = d;
    }
  };

  // ---- pipelined main loop: 1 barrier per 448-pixel iteration ----
  gen_curves(0, 0);
  __syncthreads();

  for (int it = 0; it < NIT; ++it) {
    const int cur = it & 1;
    if (it + 1 < NIT) gen_curves(it + 1, cur ^ 1);   // co-executes with WMMAs

    const _Float16* aptr0 = xh + row0 + kbase + it * PITER + ak;
    const _Float16* aptr1 = xh + row1 + kbase + it * PITER + ak;

#pragma unroll
    for (int sub = 0; sub < NSUB; ++sub) {
      union { u32x4 u[2]; v16h v; } b0, b1, a0, a1;
      const u32x4* pb0 = (const u32x4*)&sB[cur][bn][sub * 32 + bcol];
      const u32x4* pb1 = (const u32x4*)&sB[cur][16 + bn][sub * 32 + bcol];
      b0.u[0] = pb0[0]; b0.u[1] = pb0[1];
      b1.u[0] = pb1[0]; b1.u[1] = pb1[1];
      a0.u[0] = *(const u32x4*)(aptr0 + sub * 32);
      a0.u[1] = *(const u32x4*)(aptr0 + sub * 32 + 8);
      a1.u[0] = *(const u32x4*)(aptr1 + sub * 32);
      a1.u[1] = *(const u32x4*)(aptr1 + sub * 32 + 8);

      acc00 = __builtin_amdgcn_wmma_f32_16x16x32_f16(false, a0.v, false, b0.v,
                                                     (short)0, acc00, false, false);
      acc01 = __builtin_amdgcn_wmma_f32_16x16x32_f16(false, a0.v, false, b1.v,
                                                     (short)0, acc01, false, false);
      acc10 = __builtin_amdgcn_wmma_f32_16x16x32_f16(false, a1.v, false, b0.v,
                                                     (short)0, acc10, false, false);
      acc11 = __builtin_amdgcn_wmma_f32_16x16x32_f16(false, a1.v, false, b1.v,
                                                     (short)0, acc11, false, false);
    }
    __syncthreads();
  }

  // ---- scaled atomic accumulation per C-tile layout:
  //      VGPR j of lane L holds (M = j + 8*(L/16), N = L%16)
  const float scale = 1.0f / (float)KTOT;
  const int ncol0 = c0 + (lane & 15);
  const int ncol1 = ncol0 + 16;
  const int mrow  = wv * 32 + ((lane >> 4) << 3);
#pragma unroll
  for (int j = 0; j < 8; ++j) {
    size_t r0 = (size_t)(mrow + j) * CC;
    size_t r1 = (size_t)(mrow + 16 + j) * CC;
    unsafeAtomicAdd(&out[r0 + ncol0], acc00[j] * scale);
    unsafeAtomicAdd(&out[r0 + ncol1], acc01[j] * scale);
    unsafeAtomicAdd(&out[r1 + ncol0], acc10[j] * scale);
    unsafeAtomicAdd(&out[r1 + ncol1], acc11[j] * scale);
  }
}

// ---------------------------------------------------------------------------
extern "C" void kernel_launch(void* const* d_in, const int* in_sizes, int n_in,
                              void* d_out, int out_size, void* d_ws, size_t ws_size,
                              hipStream_t stream) {
  const float* x   = (const float*)d_in[0];  // (256,224,224)
  const float* pos = (const float*)d_in[1];  // (1,1,1024,2)
  const float* sg  = (const float*)d_in[2];  // (1,1,1024)
  const float* wt  = (const float*)d_in[3];  // (1,1,1024)
  // d_in[4] = xs grid, d_in[5] = ys grid: recomputed analytically on-chip
  float* out = (float*)d_out;                // (256,1024)
  _Float16* xh = (_Float16*)d_ws;            // 25.7 MB f16 copy of x

  const int nelem = BB * KTOT;               // 12,845,056
  const int n4 = nelem / 4;
  xcast_f16<<<dim3((n4 + 255) / 256), dim3(256), 0, stream>>>(x, xh, n4);
  zero_out<<<dim3((BB * CC + 255) / 256), dim3(256), 0, stream>>>(out, BB * CC);
  blob_wmma<<<dim3(NBLK * KSPLIT), dim3(256), 0, stream>>>(xh, pos, sg, wt, out);
}
